// Model_5540507811947
// MI455X (gfx1250) — matrix operand order repaired and checked by reading, not compiled, not run on hardware
//
#include <hip/hip_runtime.h>

// ---------------------------------------------------------------------------
// Shapes (fixed by the reference): B=2, L=2048, H=1024, NH=16, D=64, M=2048
// ---------------------------------------------------------------------------
#define BB   2
#define LL   2048
#define HDIM 1024
#define NHH  16
#define DDIM 64
#define MM   2048

typedef __attribute__((ext_vector_type(16))) __bf16        v16bf;
typedef __attribute__((ext_vector_type(8)))  float         v8f;
typedef __attribute__((ext_vector_type(4)))  unsigned int  v4u;
typedef __attribute__((ext_vector_type(8)))  int           v8i;
typedef __attribute__((ext_vector_type(4)))  int           v4i;

union BfPack {
    unsigned int u[8];
    v16bf v;
};

__device__ __forceinline__ unsigned short f2bf(float f) {
    unsigned int u = __float_as_uint(f);
    unsigned int r = u + 0x7FFFu + ((u >> 16) & 1u);   // RNE
    return (unsigned short)(r >> 16);
}
__device__ __forceinline__ v8f wmma_bf16(v16bf a, v16bf b, v8f c) {
    return __builtin_amdgcn_wmma_f32_16x16x32_bf16(
        false, a, false, b, (short)0, c, false, false);
}

// --- CDNA5 async global->LDS copy (ASYNCcnt-tracked), ISA 15.18.3 ----------
__device__ __forceinline__ unsigned lds_off32(const void* p) {
    return (unsigned)(unsigned long long)p;            // LDS flat addr low bits
}
__device__ __forceinline__ void async_b128(unsigned loff, const void* g) {
    asm volatile("global_load_async_to_lds_b128 %0, %1, off"
                 :: "v"(loff), "v"(g) : "memory");
}
__device__ __forceinline__ void async_wait0() {
    asm volatile("s_wait_asynccnt 0x0" ::: "memory");
}

// --- CDNA5 Tensor Data Mover (TENSORcnt-tracked), ISA ch.7/8 ---------------
#if __has_builtin(__builtin_amdgcn_tensor_load_to_lds) && __has_builtin(__builtin_amdgcn_s_wait_tensorcnt)
#define USE_TDM 1
#else
#define USE_TDM 0
#endif

#if USE_TDM
__device__ __forceinline__ void tdm_issue(v4u g0, v8i g1) {
    v4i z4 = {0, 0, 0, 0};
#if __clang_major__ >= 23
    v8i z8 = {0, 0, 0, 0, 0, 0, 0, 0};
    __builtin_amdgcn_tensor_load_to_lds(g0, g1, z4, z4, z8, 0);
#else
    __builtin_amdgcn_tensor_load_to_lds(g0, g1, z4, z4, 0);
#endif
}
// 2D bf16 tile load: tile_d0 x tile_d1 elements out of a row-major tensor
// whose contiguous dim0 has length tensor_d0 (= stride), into contiguous LDS.
__device__ __forceinline__ void tdm_load_tile2d(
    unsigned lds_addr, const void* gaddr,
    unsigned tensor_d0, unsigned tensor_d1,
    unsigned tile_d0, unsigned tile_d1, unsigned stride_d0)
{
    const unsigned long long ga = (unsigned long long)gaddr;
    v4u g0;
    g0[0] = 1u;                                             // count=1 (valid D#)
    g0[1] = lds_addr;                                       // LDS byte address
    g0[2] = (unsigned)ga;                                   // global addr [31:0]
    g0[3] = (unsigned)((ga >> 32) & 0x01FFFFFFu) | (2u << 30); // [56:32] | type=2
    v8i g1;
    g1[0] = (int)(1u << 16);                                // data_size=1 (2 bytes)
    g1[1] = (int)((tensor_d0 & 0xFFFFu) << 16);             // tensor_dim0[15:0]
    g1[2] = (int)((tensor_d0 >> 16) | ((tensor_d1 & 0xFFFFu) << 16));
    g1[3] = (int)((tensor_d1 >> 16) | (tile_d0 << 16));     // tile_dim0
    g1[4] = (int)(tile_d1 & 0xFFFFu);                       // tile_dim1, tile_dim2=0
    g1[5] = (int)stride_d0;                                 // dim0_stride[31:0]
    g1[6] = 0;                                              // stride hi, dim1_stride lo
    g1[7] = 0;
    tdm_issue(g0, g1);
}
#endif

// ---------------------------------------------------------------------------
// Kernel 0: one-shot bf16 conversions so hot loops do pure b32 loads.
// ---------------------------------------------------------------------------
__global__ __launch_bounds__(256) void prep_kernel(
    const float* __restrict__ X,
    const float* __restrict__ Wq, const float* __restrict__ Wk,
    const float* __restrict__ Wv, const float* __restrict__ Emb,
    unsigned short* __restrict__ Xb,
    unsigned short* __restrict__ Wt,
    unsigned short* __restrict__ Eb)
{
    const int seg = blockIdx.y;
    const int stride = gridDim.x * 256;
    const int idx0 = blockIdx.x * 256 + threadIdx.x;
    if (seg == 0) {
        for (int i = idx0; i < 4096 * 1024; i += stride) Xb[i] = f2bf(X[i]);
    } else if (seg <= 3) {
        const float* W = seg == 1 ? Wq : (seg == 2 ? Wk : Wv);
        unsigned short* T = Wt + (size_t)(seg - 1) * 1024 * 1024;
        for (int i = idx0; i < 1024 * 1024; i += stride) {
            const int n = i >> 10, k = i & 1023;
            T[i] = f2bf(W[(size_t)k * 1024 + n]);           // transpose
        }
    } else {
        for (int i = idx0; i < (2 * MM - 1) * DDIM; i += stride) Eb[i] = f2bf(Emb[i]);
    }
}

// ---------------------------------------------------------------------------
// Kernel 1: QKV projection, bf16 WMMA GEMM, 32x64 block per wave (2x4 tiles).
//   Q,K -> [B,NH,L,D] bf16 ; V -> [B,NH,D,L] bf16 (transposed for P@V)
// ---------------------------------------------------------------------------
__global__ __launch_bounds__(128) void qkv_kernel(
    const unsigned short* __restrict__ Xb,
    const unsigned short* __restrict__ Wt,
    const float* __restrict__ bq, const float* __restrict__ bk,
    const float* __restrict__ bv,
    unsigned short* __restrict__ Qb,
    unsigned short* __restrict__ Kb,
    unsigned short* __restrict__ Vt)
{
    const int which = blockIdx.y;                     // 0=Q 1=K 2=V
    const float* bias = which == 0 ? bq : (which == 1 ? bk : bv);
    const unsigned short* W = Wt + (size_t)which * 1024 * 1024;

    const int lane = threadIdx.x & 31;
    const int wv   = threadIdx.x >> 5;
    const int tile = blockIdx.x * 4 + wv;             // 0..2047
    const int mt = tile >> 4;                         // 0..127 (32-row blocks)
    const int nt = tile & 15;                         // 0..15  (64-col blocks)
    const int hf = lane >> 4, ln = lane & 15;

    const unsigned int* xr[2];
    const unsigned int* wr[4];
#pragma unroll
    for (int ms = 0; ms < 2; ++ms)
        xr[ms] = (const unsigned int*)(Xb + (size_t)(mt * 32 + ms * 16 + ln) * 1024);
#pragma unroll
    for (int ns = 0; ns < 4; ++ns)
        wr[ns] = (const unsigned int*)(W + (size_t)(nt * 64 + ns * 16 + ln) * 1024);

    v8f acc[8];
#pragma unroll
    for (int i = 0; i < 8; ++i) acc[i] = (v8f){};

    for (int kc = 0; kc < 512; kc += 16) {            // kc counts bf16 pairs
        v16bf a[2], b[4];
#pragma unroll
        for (int ms = 0; ms < 2; ++ms) {              // A-layout: K split per half
            BfPack t; const int kb = kc + hf * 4;
#pragma unroll
            for (int i = 0; i < 4; ++i) { t.u[i] = xr[ms][kb + i]; t.u[4 + i] = xr[ms][kb + 8 + i]; }
            a[ms] = t.v;
        }
#pragma unroll
        for (int ns = 0; ns < 4; ++ns) {              // B-layout: K split per half
            BfPack t; const int kb = kc + hf * 4;
#pragma unroll
            for (int i = 0; i < 4; ++i) { t.u[i] = wr[ns][kb + i]; t.u[4 + i] = wr[ns][kb + 8 + i]; }
            b[ns] = t.v;
        }
#pragma unroll
        for (int ms = 0; ms < 2; ++ms)
#pragma unroll
            for (int ns = 0; ns < 4; ++ns)
                acc[ms * 4 + ns] = wmma_bf16(a[ms], b[ns], acc[ms * 4 + ns]);
    }

    // Branchless epilogue: one pointer + affine layout select, no per-element
    // control flow.  Q/K: idx = bh*(L*D) + l*D + d ;  V: idx = bh*(D*L) + d*L + l
    unsigned short* dst = which == 0 ? Qb : (which == 1 ? Kb : Vt);
    const size_t lmul = (which == 2) ? (size_t)1 : (size_t)DDIM;
    const size_t dmul = (which == 2) ? (size_t)LL : (size_t)1;
#pragma unroll
    for (int ms = 0; ms < 2; ++ms) {
#pragma unroll
        for (int ns = 0; ns < 4; ++ns) {
            const int ncol = nt * 64 + ns * 16 + ln;
            const float bval = bias[ncol];
            const int hd = ncol >> 6, dd = ncol & 63;
#pragma unroll
            for (int v = 0; v < 8; ++v) {
                const int grow = mt * 32 + ms * 16 + v + 8 * hf;
                const int bidx = grow >> 11, l = grow & (LL - 1);
                dst[(size_t)(bidx * NHH + hd) * (LL * DDIM)
                    + (size_t)l * lmul + (size_t)dd * dmul]
                    = f2bf(acc[ms * 4 + ns][v] + bval);
            }
        }
    }
}

// ---------------------------------------------------------------------------
// Kernel 2: flash attention + relative_key_query.  4 waves/block share one
// (b,h); K tiles staged by the Tensor Data Mover (TENSORcnt), V tiles by
// per-lane async global->LDS copies (ASYNCcnt), both double-buffered.
// Per wave, per 16-col step:
//   S  = Q @ K^T (2 wmma), Cq = Q @ Ehat^T (4), Ck = K @ Ehat^T (4),
//   diagonal gathers qpe[i,j]=Cq[i,15+i-j], kpe[i,j]=Ck[j,15+i-j],
//   online softmax; every 2 steps O += P(16x32) @ V(32x64) (4 wmma).
// ---------------------------------------------------------------------------
__global__ __launch_bounds__(128) void attn_kernel(
    const unsigned short* __restrict__ Qb,
    const unsigned short* __restrict__ Kb,
    const unsigned short* __restrict__ Vt,
    const unsigned short* __restrict__ Eb,
    const float* __restrict__ mask,
    float* __restrict__ out)
{
    __shared__ unsigned short kbuf[2][16][64];        // K tile, double-buffered
    __shared__ unsigned short vbuf[2][64][32];        // V tile [d][l], per pair
    __shared__ float cqlds[4][16][32];
    __shared__ float cklds[4][16][32];
    __shared__ float plds [4][16][32];
    __shared__ float alds [4][16];

    const int tid  = threadIdx.x;
    const int lane = tid & 31;
    const int wv   = tid >> 5;
    const int gw   = blockIdx.x * 4 + wv;             // 4 consecutive strips
    const int strip = gw & 127;
    const int bh    = gw >> 7;                        // same for all 4 waves
    const int bidx  = bh >> 4, hd = bh & 15;
    const int ls    = strip * 16;
    const int hf = lane >> 4, ln = lane & 15;

    const unsigned short* Qh = Qb + (size_t)bh * LL * DDIM;
    const unsigned short* Kh = Kb + (size_t)bh * LL * DDIM;
    const unsigned short* Vh = Vt + (size_t)bh * DDIM * LL;

#if USE_TDM
    // K tile: 16 rows x 64 bf16 (contiguous rows of Kh) via one TDM descriptor
    auto stage_k = [&](int buf, int r0) {
        if (wv == 0)
            tdm_load_tile2d(lds_off32(&kbuf[buf][0][0]),
                            Kh + (size_t)r0 * DDIM,
                            /*tensor_d0=*/DDIM, /*tensor_d1=*/LL,
                            /*tile_d0=*/DDIM, /*tile_d1=*/16,
                            /*stride_d0=*/DDIM);
    };
#else
    // fallback: 128 threads x 16B = one 2KB K tile per issue
    auto stage_k = [&](int buf, int r0) {
        const char* g = (const char*)(Kh + (size_t)(r0 + (tid >> 3)) * DDIM) + (tid & 7) * 16;
        async_b128(lds_off32(&kbuf[buf][0][0]) + tid * 16, g);
    };
#endif
    // V tile: 64 rows x 64B = 4KB: 256 chunks, 2 per thread (async path)
    auto stage_v = [&](int buf, int rp0) {
#pragma unroll
        for (int r = 0; r < 2; ++r) {
            const int c = tid + r * 128;
            const char* g = (const char*)(Vh + (size_t)(c >> 2) * LL + rp0) + (c & 3) * 16;
            async_b128(lds_off32(&vbuf[buf][0][0]) + c * 16, g);
        }
    };
    auto stage_wait = [&]() {
#if USE_TDM
        if (wv == 0) __builtin_amdgcn_s_wait_tensorcnt((short)0);
#endif
        async_wait0();
        __syncthreads();
    };

    // ---- persistent A-operand: this strip's Q (16x64) ----
    v16bf aq[2];
    {
        const unsigned int* qrow = (const unsigned int*)(Qh + (size_t)(ls + ln) * DDIM);
#pragma unroll
        for (int kc = 0; kc < 2; ++kc) {
            BfPack a; const int kb = kc * 16 + hf * 4;
#pragma unroll
            for (int i = 0; i < 4; ++i) { a.u[i] = qrow[kb + i]; a.u[4 + i] = qrow[kb + 8 + i]; }
            aq[kc] = a.v;
        }
    }

    v8f o0 = {}, o1 = {}, o2 = {}, o3 = {};
    float m_run[8], l_run[8];
#pragma unroll
    for (int v = 0; v < 8; ++v) { m_run[v] = -1e30f; l_run[v] = 0.f; }

    stage_k(0, 0);
    stage_v(0, 0);
    stage_wait();

    for (int step = 0; step < LL / 16; ++step) {
        const int r0 = step * 16;
        const int b  = step & 1;

        // prefetch next tiles into the other buffers (overlaps compute)
        if (step + 1 < LL / 16) stage_k(b ^ 1, r0 + 16);
        if (!(step & 1)) {
            const int p = step >> 1;
            if (p + 1 < LL / 32) stage_v((p + 1) & 1, (p + 1) * 32);
        }

        // ---- B-operand: K tile transposed (from LDS) ----
        const unsigned int* kr = (const unsigned int*)&kbuf[b][ln][0];
        v16bf bkk[2];
#pragma unroll
        for (int kc = 0; kc < 2; ++kc) {
            BfPack t; const int base = kc * 16 + hf * 4;
#pragma unroll
            for (int i = 0; i < 4; ++i) { t.u[i] = kr[base + i]; t.u[4 + i] = kr[base + 8 + i]; }
            bkk[kc] = t.v;
        }
        v8f s = {};
        s = wmma_bf16(aq[0], bkk[0], s);
        s = wmma_bf16(aq[1], bkk[1], s);

        // ---- B-operand: distance-embedding window Ehat^T (bf16) ----
        const int ebase = ls - r0 + (MM - 1) - 15;
        v16bf be[2][2];
#pragma unroll
        for (int ct = 0; ct < 2; ++ct) {
            int er = ebase + ct * 16 + ln;
            er = er < 0 ? 0 : (er > 2 * MM - 2 ? 2 * MM - 2 : er);
            const unsigned int* erp = (const unsigned int*)(Eb + (size_t)er * DDIM);
#pragma unroll
            for (int kc = 0; kc < 2; ++kc) {
                BfPack t; const int kb = kc * 16 + hf * 4;
#pragma unroll
                for (int i = 0; i < 4; ++i) { t.u[i] = erp[kb + i]; t.u[4 + i] = erp[kb + 8 + i]; }
                be[kc][ct] = t.v;
            }
        }

        v8f cq0 = {}, cq1 = {}, ck0 = {}, ck1 = {};
        cq0 = wmma_bf16(aq[0], be[0][0], cq0);
        cq0 = wmma_bf16(aq[1], be[1][0], cq0);
        cq1 = wmma_bf16(aq[0], be[0][1], cq1);
        cq1 = wmma_bf16(aq[1], be[1][1], cq1);

        // ---- A-operand: K tile rows (from LDS) ----
        v16bf ak[2];
#pragma unroll
        for (int kc = 0; kc < 2; ++kc) {
            BfPack t; const int kb = kc * 16 + hf * 4;
#pragma unroll
            for (int i = 0; i < 4; ++i) { t.u[i] = kr[kb + i]; t.u[4 + i] = kr[kb + 8 + i]; }
            ak[kc] = t.v;
        }
        ck0 = wmma_bf16(ak[0], be[0][0], ck0);
        ck0 = wmma_bf16(ak[1], be[1][0], ck0);
        ck1 = wmma_bf16(ak[0], be[0][1], ck1);
        ck1 = wmma_bf16(ak[1], be[1][1], ck1);

        // ---- stage Cq/Ck for the diagonal gather ----
#pragma unroll
        for (int v = 0; v < 8; ++v) {
            const int i = v + 8 * hf;
            cqlds[wv][i][ln]      = cq0[v];
            cqlds[wv][i][16 + ln] = cq1[v];
            cklds[wv][i][ln]      = ck0[v];
            cklds[wv][i][16 + ln] = ck1[v];
        }

        const float maskv = mask[bidx * LL + r0 + ln];

        float t8[8];
#pragma unroll
        for (int v = 0; v < 8; ++v) {
            const int i = v + 8 * hf;
            const int c = 15 + i - ln;                // in [0,30]
            t8[v] = (s[v] + cqlds[wv][i][c] + cklds[wv][ln][c]) * 0.125f + maskv;
        }
#pragma unroll
        for (int v = 0; v < 8; ++v) {
            float cm = t8[v];
#pragma unroll
            for (int off = 1; off < 16; off <<= 1)
                cm = fmaxf(cm, __shfl_xor(cm, off, 32));
            const float mn = fmaxf(m_run[v], cm);
            const float al = __expf(m_run[v] - mn);
            const float p  = __expf(t8[v] - mn);
            float rs = p;
#pragma unroll
            for (int off = 1; off < 16; off <<= 1)
                rs += __shfl_xor(rs, off, 32);
            l_run[v] = l_run[v] * al + rs;
            m_run[v] = mn;
            o0[v] *= al; o1[v] *= al; o2[v] *= al; o3[v] *= al;
            const int i = v + 8 * hf;
            plds[wv][i][(step & 1) * 16 + ln] = p;
            alds[wv][i] = al;
        }

        // ---- every 2 steps: O += P(16x32) @ V(32x64) ----
        if (step & 1) {
            const float af = alds[wv][ln];            // rescale deferred P half
            BfPack ap; const int kb = hf * 8;
#pragma unroll
            for (int i = 0; i < 4; ++i) {
                ap.u[i] = ((unsigned)f2bf(plds[wv][ln][kb + 2 * i] * af)) |
                          ((unsigned)f2bf(plds[wv][ln][kb + 2 * i + 1] * af) << 16);
                ap.u[4 + i] = ((unsigned)f2bf(plds[wv][ln][kb + 16 + 2 * i])) |
                              ((unsigned)f2bf(plds[wv][ln][kb + 16 + 2 * i + 1]) << 16);
            }
            const int pb = (step >> 1) & 1;
#pragma unroll
            for (int ct = 0; ct < 4; ++ct) {
                const unsigned int* vr = (const unsigned int*)&vbuf[pb][ct * 16 + ln][0];
                BfPack bvv;
#pragma unroll
                for (int i = 0; i < 4; ++i) { bvv.u[i] = vr[hf * 4 + i]; bvv.u[4 + i] = vr[hf * 4 + 8 + i]; }
                v8f* oP = ct == 0 ? &o0 : ct == 1 ? &o1 : ct == 2 ? &o2 : &o3;
                *oP = wmma_bf16(ap.v, bvv.v, *oP);
            }
        }

        stage_wait();                                 // prefetched tiles ready
    }

    // ---- normalize and write ctx -> [B, L, NH, D] ----
#pragma unroll
    for (int v = 0; v < 8; ++v) {
        const int i = v + 8 * hf;
        const int l = ls + i;
        const float inv = 1.0f / l_run[v];
        const size_t base = ((size_t)(bidx * LL + l) * NHH + hd) * DDIM;
        out[base +  0 + ln] = o0[v] * inv;
        out[base + 16 + ln] = o1[v] * inv;
        out[base + 32 + ln] = o2[v] * inv;
        out[base + 48 + ln] = o3[v] * inv;
    }
}

// ---------------------------------------------------------------------------
extern "C" void kernel_launch(void* const* d_in, const int* in_sizes, int n_in,
                              void* d_out, int out_size, void* d_ws, size_t ws_size,
                              hipStream_t stream)
{
    const float* X    = (const float*)d_in[0];
    const float* mask = (const float*)d_in[1];
    const float* Wq   = (const float*)d_in[2];
    const float* bq   = (const float*)d_in[3];
    const float* Wk   = (const float*)d_in[4];
    const float* bk   = (const float*)d_in[5];
    const float* Wv   = (const float*)d_in[6];
    const float* bv   = (const float*)d_in[7];
    const float* Emb  = (const float*)d_in[8];
    float* out = (float*)d_out;

    // workspace (bf16 elements)
    unsigned short* Xb = (unsigned short*)d_ws;                 // 4096*1024
    unsigned short* Wt = Xb + (size_t)4096 * 1024;              // 3*1024*1024
    unsigned short* Eb = Wt + (size_t)3 * 1024 * 1024;          // 4096*64 (pad)
    unsigned short* Qb = Eb + (size_t)4096 * 64;                // B*NH*L*D
    unsigned short* Kb = Qb + (size_t)BB * NHH * LL * DDIM;
    unsigned short* Vt = Kb + (size_t)BB * NHH * LL * DDIM;

    prep_kernel<<<dim3(2048, 5, 1), dim3(256, 1, 1), 0, stream>>>(
        X, Wq, Wk, Wv, Emb, Xb, Wt, Eb);

    qkv_kernel<<<dim3(512, 3, 1), dim3(128, 1, 1), 0, stream>>>(
        Xb, Wt, bq, bk, bv, Qb, Kb, Vt);

    attn_kernel<<<dim3(1024, 1, 1), dim3(128, 1, 1), 0, stream>>>(
        Qb, Kb, Vt, Eb, mask, out);
}
